// BranchFusSSM_20134806684288
// MI455X (gfx1250) — compile-verified, matrix operand order checked by reading, not run on hardware
//
#include <hip/hip_runtime.h>
#include <hip/hip_bf16.h>

// ---------------- problem constants (from reference) ----------------
#define BSZ     2
#define DM      96            // D_MODEL
#define HH      128
#define WW      128
#define LL      (HH * WW)     // 16384
#define DI      96            // D_INNER
#define DSTATE  16
#define KDIR    4
#define DTR     6             // DT_RANK
#define C38     (DTR + 2 * DSTATE)   // 38
#define ZST     128           // Z rows: [0,96)=delta_pre, [96,112)=B, [112,128)=C
#define LNEPS   1e-5f

typedef __attribute__((ext_vector_type(16))) __bf16 v16bf;
typedef __attribute__((ext_vector_type(8)))  float  v8f;
typedef __attribute__((ext_vector_type(4)))  unsigned int v4u;
typedef __attribute__((ext_vector_type(4)))  int    v4i;
typedef __attribute__((ext_vector_type(8)))  int    v8i;

#if defined(__has_builtin)
#if __has_builtin(__builtin_amdgcn_tensor_load_to_lds)
#define USE_TDM 1
#endif
#endif
#ifndef USE_TDM
#define USE_TDM 0
#endif

// K index (within a 32-wide chunk) held by fragment element e for a lane with
// half = lane>>4, per the CDNA5 16-bit A/B fragment layout (ISA 7.12.2).
__device__ __forceinline__ int kmap(int e, int half) {
    return (e & 7) + half * 8 + ((e >> 3) << 4);
}

#if USE_TDM
// Issue a TDM 2-D tile load (f32 elements) Global -> LDS per ISA ch.8 D# layout.
//   group0: count=1 | lds_addr | global_addr[56:0] | type=2
//   group1: data_size=4B, tensor_dim0/1, tile_dim0/1, tensor_dim0_stride
__device__ __forceinline__ void tdm_load_2d(unsigned int lds_off, const void* gaddr,
                                            unsigned int width, unsigned int height,
                                            unsigned int stride) {
    const unsigned long long ga = (unsigned long long)(uintptr_t)gaddr;
    v4u g0;
    g0[0] = 1u;                                        // count=1 (valid user D#)
    g0[1] = lds_off;                                   // lds_addr (bytes)
    g0[2] = (unsigned int)ga;                          // global_addr[31:0]
    g0[3] = (unsigned int)(ga >> 32) | (2u << 30);     // global_addr[56:32] | type=2
    v8i g1;
    g1[0] = (int)(2u << 16);                           // wg_mask=0, data_size=2 (4B)
    g1[1] = (int)(width << 16);                        // tensor_dim0[15:0] @bits63:48
    g1[2] = (int)((width >> 16) | (height << 16));     // dim0 hi | tensor_dim1 lo
    g1[3] = (int)((height >> 16) | (width << 16));     // dim1 hi | tile_dim0
    g1[4] = (int)height;                               // tile_dim1 (tile_dim2=0)
    g1[5] = (int)stride;                               // tensor_dim0_stride[31:0]
    g1[6] = 0;
    g1[7] = 0;
    v4i gz = {0, 0, 0, 0};
#if __clang_major__ >= 23
    v8i gz8 = {0, 0, 0, 0, 0, 0, 0, 0};
    __builtin_amdgcn_tensor_load_to_lds(g0, g1, gz, gz, gz8, 0);
#else
    __builtin_amdgcn_tensor_load_to_lds(g0, g1, gz, gz, 0);
#endif
}
#endif

// ---------------- K0: fold weights ----------------
// Wbig[k] (128x96): rows 0..95  = dtw_k (96x6) @ xpw_k[0:6]  (delta pre-proj)
//                   rows 96..127= xpw_k[6:38]                 (B then C rows)
// Dsum[d] = sum_k Ds[k*DI+d]
__global__ __launch_bounds__(128)
void prep_kernel(const float* __restrict__ xpw, const float* __restrict__ dtw,
                 const float* __restrict__ Ds, float* __restrict__ Wbig,
                 float* __restrict__ Dsum) {
    const int k = blockIdx.x;
    const int r = threadIdx.x;           // 0..127
    for (int c = 0; c < DM; ++c) {
        float v;
        if (r < DI) {
            v = 0.f;
            #pragma unroll
            for (int j = 0; j < DTR; ++j)
                v += dtw[(k * DI + r) * DTR + j] * xpw[(k * C38 + j) * DI + c];
        } else {
            v = xpw[(k * C38 + (DTR + r - DI)) * DI + c];
        }
        Wbig[(k * ZST + r) * DM + c] = v;
    }
    if (k == 0 && r < DI) {
        float s = 0.f;
        #pragma unroll
        for (int kk = 0; kk < KDIR; ++kk) s += Ds[kk * DI + r];
        Dsum[r] = s;
    }
}

// ---------------- K1: input projections (WMMA) ----------------
// XT[b][l][d] = sum_c Wx[d,c] * x[b,c,l]   (and YT from Wy,y)
__global__ __launch_bounds__(32)
void proj_in_kernel(const float* __restrict__ x, const float* __restrict__ y,
                    const float* __restrict__ Wx, const float* __restrict__ Wy,
                    float* __restrict__ XT, float* __restrict__ YT) {
    const int lane = threadIdx.x;
    const int n    = lane & 15;
    const int half = lane >> 4;
    const int l0 = blockIdx.x * 16;
    const int d0 = blockIdx.y * 16;
    const int b  = blockIdx.z >> 1;
    const int which = blockIdx.z & 1;
    const float* W   = which ? Wy : Wx;
    const float* in  = (which ? y : x) + (size_t)b * DM * LL;
    float*       out = (which ? YT : XT) + (size_t)b * LL * DI;

    v8f acc = {};
    #pragma unroll
    for (int kk = 0; kk < DM; kk += 32) {
        v16bf af, bfv;
        #pragma unroll
        for (int e = 0; e < 16; ++e) {
            const int k = kk + kmap(e, half);
            af[e]  = (__bf16)W[(d0 + n) * DM + k];            // A: row M=d
            bfv[e] = (__bf16)in[(size_t)k * LL + (l0 + n)];   // B: col N=l
        }
        acc = __builtin_amdgcn_wmma_f32_16x16x32_bf16(
                  false, af, false, bfv, (short)0, acc, false, false);
    }
    #pragma unroll
    for (int v = 0; v < 8; ++v) {
        const int m = v + 8 * half;                           // d row
        out[(size_t)(l0 + n) * DI + (d0 + m)] = acc[v];       // transposed store
    }
}

// ---------------- K2: Z = Wbig_k @ x_in (WMMA + TDM), stored [l][128] -------
// One 256-thread block per 16-wide l-tile. TDM stages Wbig_k (128x96 f32) and
// the XT tile (16x96 f32) into LDS once; 8 waves each compute one 16-row
// r-tile from LDS fragments.
__global__ __launch_bounds__(256)
void proj_z_kernel(const float* __restrict__ XT, const float* __restrict__ Wbig,
                   float* __restrict__ Z) {
    __shared__ float lWB[ZST * DM];   // 48 KB
    __shared__ float lXT[16 * DM];    // 6 KB
    const int tid  = threadIdx.x;
    const int lane = tid & 31;
    const int wave = tid >> 5;                 // 0..7 -> r-tile
    const int n    = lane & 15;
    const int half = lane >> 4;
    const int l0 = blockIdx.x * 16;
    const int bk = blockIdx.y;                 // b*4 + k
    const int b  = bk >> 2, k = bk & 3;
    const float* xt = XT + (size_t)b * LL * DI;
    const float* wb = Wbig + (size_t)k * ZST * DM;

#if USE_TDM
    if (tid < 32) {                            // wave 0 issues both DMA tiles
        tdm_load_2d((unsigned int)(uintptr_t)lWB, wb, DM, ZST, DM);
        tdm_load_2d((unsigned int)(uintptr_t)lXT, xt + (size_t)l0 * DI, DI, 16, DI);
        __builtin_amdgcn_s_wait_tensorcnt(0);
    }
#else
    for (int i = tid; i < ZST * DM; i += 256) lWB[i] = wb[i];
    for (int i = tid; i < 16 * DM; i += 256) lXT[i] = xt[(size_t)l0 * DI + i];
#endif
    __syncthreads();

    const int r0 = wave * 16;
    v8f acc = {};
    #pragma unroll
    for (int kk = 0; kk < DM; kk += 32) {
        v16bf af, bfv;
        #pragma unroll
        for (int e = 0; e < 16; ++e) {
            const int kc = kk + kmap(e, half);
            af[e]  = (__bf16)lXT[n * DM + kc];        // A: row M=l (from LDS)
            bfv[e] = (__bf16)lWB[(r0 + n) * DM + kc]; // B: col N=r (from LDS)
        }
        acc = __builtin_amdgcn_wmma_f32_16x16x32_bf16(
                  false, af, false, bfv, (short)0, acc, false, false);
    }
    float* zo = Z + (size_t)bk * LL * ZST;
    #pragma unroll
    for (int v = 0; v < 8; ++v) {
        const int m = v + 8 * half;                           // l row
        zo[(size_t)(l0 + m) * ZST + (r0 + n)] = acc[v];
    }
}

// ---------------- K4: selective scan ----------------
// One block per (b,k); lane = channel d. State h[16] in registers.
__global__ __launch_bounds__(96)
void scan_kernel(const float* __restrict__ Z, const float* __restrict__ YT,
                 const float* __restrict__ A_logs, const float* __restrict__ dt_bias,
                 float* __restrict__ YACC) {
    const int bk = blockIdx.x;
    const int b  = bk >> 2, k = bk & 3;
    const int d  = threadIdx.x;            // 0..95

    float An[DSTATE];
    #pragma unroll
    for (int nn = 0; nn < DSTATE; ++nn)
        An[nn] = -__expf(A_logs[(k * DI + d) * DSTATE + nn]);
    const float bias = dt_bias[k * DI + d];

    const float* zb = Z  + (size_t)bk * LL * ZST;
    const float* yt = YT + (size_t)b  * LL * DI;
    float*       ya = YACC + (size_t)b * LL * DI;

    float h[DSTATE];
    #pragma unroll
    for (int nn = 0; nn < DSTATE; ++nn) h[nn] = 0.f;

    for (int t = 0; t < LL; ++t) {
        const int tt = (k & 2) ? (LL - 1 - t) : t;
        const int l  = (k & 1) ? (((tt & (HH - 1)) << 7) | (tt >> 7)) : tt;
        const float* zrow = zb + (size_t)l * ZST;

        // prefetch the row needed 8 steps ahead (global_prefetch_b8)
        const int tf = t + 8;
        if (tf < LL) {
            const int ttf = (k & 2) ? (LL - 1 - tf) : tf;
            const int lf  = (k & 1) ? (((ttf & (HH - 1)) << 7) | (ttf >> 7)) : ttf;
            __builtin_prefetch(zb + (size_t)lf * ZST, 0, 1);
        }

        const float dp    = zrow[d] + bias;
        const float delta = (dp > 20.f) ? dp : __logf(1.f + __expf(dp));
        const float u     = yt[(size_t)l * DI + d];
        const float du    = delta * u;

        float yv = 0.f;
        #pragma unroll
        for (int nn = 0; nn < DSTATE; ++nn) {
            const float dA = __expf(delta * An[nn]);
            const float hn = h[nn] * dA + du * zrow[DI + nn];
            h[nn] = hn;
            yv += hn * zrow[DI + DSTATE + nn];
        }
        atomicAdd(&ya[(size_t)l * DI + d], yv);   // global_atomic_add_f32
    }
}

// ---------------- K5: skip term + LayerNorm (in place on YACC) ----------------
__global__ __launch_bounds__(96)
void ln_kernel(float* __restrict__ YACC, const float* __restrict__ YT,
               const float* __restrict__ Dsum, const float* __restrict__ gamma,
               const float* __restrict__ beta) {
    __shared__ float red[DI];
    const size_t row = blockIdx.x;                // b*L + l
    const int d = threadIdx.x;
    const float v = YACC[row * DI + d] + Dsum[d] * YT[row * DI + d];
    red[d] = v;
    __syncthreads();
    float s = 0.f;
    #pragma unroll 8
    for (int i = 0; i < DI; ++i) s += red[i];
    const float mu = s * (1.f / DI);
    __syncthreads();
    red[d] = (v - mu) * (v - mu);
    __syncthreads();
    float s2 = 0.f;
    #pragma unroll 8
    for (int i = 0; i < DI; ++i) s2 += red[i];
    const float var = s2 * (1.f / DI);
    YACC[row * DI + d] = (v - mu) * rsqrtf(var + LNEPS) * gamma[d] + beta[d];
}

// ---------------- K6: output projection (WMMA) ----------------
// out[b][c][l] = sum_d Wout[c,d] * YM[b][l][d]
__global__ __launch_bounds__(32)
void proj_out_kernel(const float* __restrict__ YM, const float* __restrict__ Wout,
                     float* __restrict__ out) {
    const int lane = threadIdx.x;
    const int n    = lane & 15;
    const int half = lane >> 4;
    const int l0 = blockIdx.x * 16;
    const int c0 = blockIdx.y * 16;
    const int b  = blockIdx.z;
    const float* ym = YM + (size_t)b * LL * DI;

    v8f acc = {};
    #pragma unroll
    for (int kk = 0; kk < DI; kk += 32) {
        v16bf af, bfv;
        #pragma unroll
        for (int e = 0; e < 16; ++e) {
            const int kd = kk + kmap(e, half);
            af[e]  = (__bf16)Wout[(c0 + n) * DI + kd];           // A: row M=c
            bfv[e] = (__bf16)ym[(size_t)(l0 + n) * DI + kd];     // B: col N=l
        }
        acc = __builtin_amdgcn_wmma_f32_16x16x32_bf16(
                  false, af, false, bfv, (short)0, acc, false, false);
    }
    float* ob = out + (size_t)b * DM * LL;
    #pragma unroll
    for (int v = 0; v < 8; ++v) {
        const int m = v + 8 * half;                              // c row
        ob[(size_t)(c0 + m) * LL + (l0 + n)] = acc[v];
    }
}

// ---------------- host launcher ----------------
extern "C" void kernel_launch(void* const* d_in, const int* in_sizes, int n_in,
                              void* d_out, int out_size, void* d_ws, size_t ws_size,
                              hipStream_t stream) {
    (void)in_sizes; (void)n_in; (void)out_size; (void)ws_size;
    const float* x    = (const float*)d_in[0];
    const float* y    = (const float*)d_in[1];
    const float* Wx   = (const float*)d_in[2];
    const float* Wy   = (const float*)d_in[3];
    const float* xpw  = (const float*)d_in[4];
    const float* dtw  = (const float*)d_in[5];
    const float* dtb  = (const float*)d_in[6];
    const float* Alog = (const float*)d_in[7];
    const float* Ds   = (const float*)d_in[8];
    const float* lng  = (const float*)d_in[9];
    const float* lnb  = (const float*)d_in[10];
    const float* Wout = (const float*)d_in[11];
    float* out = (float*)d_out;

    // workspace carve-up (floats); total ~105 MB -> L2-resident working set
    float* ws = (float*)d_ws;
    size_t off = 0;
    float* XT = ws + off;  off += (size_t)BSZ * LL * DI;          // 12.6 MB
    float* YT = ws + off;  off += (size_t)BSZ * LL * DI;          // 12.6 MB
    float* Z  = ws + off;  off += (size_t)BSZ * KDIR * LL * ZST;  // 64 MB
    float* WB = ws + off;  off += (size_t)KDIR * ZST * DM;
    float* DS_ = ws + off; off += 128;
    float* YA = ws + off;  off += (size_t)BSZ * LL * DI;          // 12.6 MB

    prep_kernel<<<dim3(KDIR), dim3(128), 0, stream>>>(xpw, dtw, Ds, WB, DS_);
    proj_in_kernel<<<dim3(LL / 16, DI / 16, BSZ * 2), dim3(32), 0, stream>>>(
        x, y, Wx, Wy, XT, YT);
    proj_z_kernel<<<dim3(LL / 16, BSZ * KDIR), dim3(256), 0, stream>>>(XT, WB, Z);
    hipMemsetAsync(YA, 0, (size_t)BSZ * LL * DI * sizeof(float), stream);
    scan_kernel<<<dim3(BSZ * KDIR), dim3(96), 0, stream>>>(Z, YT, Alog, dtb, YA);
    ln_kernel<<<dim3(BSZ * LL), dim3(96), 0, stream>>>(YA, YT, DS_, lng, lnb);
    proj_out_kernel<<<dim3(LL / 16, DM / 16, BSZ), dim3(32), 0, stream>>>(
        YA, Wout, out);
}
